// Char_level_bidirectional_20091857011042
// MI455X (gfx1250) — matrix operand order repaired and checked by reading, not compiled, not run on hardware
//
#include <hip/hip_runtime.h>
#include <cstdint>

typedef __attribute__((ext_vector_type(16))) _Float16 v16h;
typedef __attribute__((ext_vector_type(8)))  float    v8f;
typedef __attribute__((ext_vector_type(4)))  unsigned int v4u;
typedef __attribute__((ext_vector_type(8)))  int v8i;
typedef __attribute__((ext_vector_type(4)))  int v4i;

#define VOCAB 128
#define EMB   128
#define UNITS 256
#define GATES 1024
#define SEQ   128
#define MT    16     // batch rows per workgroup (one WMMA M tile)
#define HP    264    // padded LDS h-row length in halfs (528B lane stride, conflict-free)
#define NSTAGE_FRAGS 256   // fragments of Wh staged in LDS (gates i,f = 256 KB)

// fast activations: prefer the CDNA5 hardware tanh transcendental
#if __has_builtin(__builtin_amdgcn_tanhf)
__device__ __forceinline__ float tanh_fast(float x) { return __builtin_amdgcn_tanhf(x); }
__device__ __forceinline__ float sig_fast(float x)  { return fmaf(0.5f, __builtin_amdgcn_tanhf(0.5f * x), 0.5f); }
#elif __has_builtin(__builtin_amdgcn_tanh_f32)
__device__ __forceinline__ float tanh_fast(float x) { return __builtin_amdgcn_tanh_f32(x); }
__device__ __forceinline__ float sig_fast(float x)  { return fmaf(0.5f, __builtin_amdgcn_tanh_f32(0.5f * x), 0.5f); }
#else
__device__ __forceinline__ float tanh_fast(float x) {
    return fmaf(-2.0f, __builtin_amdgcn_rcpf(__expf(2.0f * x) + 1.0f), 1.0f);
}
__device__ __forceinline__ float sig_fast(float x) {
    return __builtin_amdgcn_rcpf(1.0f + __expf(-x));
}
#endif

// global (address_space(1)) pointers keep the laundered weight stream on the
// GLOBAL vmem path (global_load_b128), not FLAT.
typedef _Float16 __attribute__((address_space(1))) gf16;
typedef v16h     __attribute__((address_space(1))) gv16h;

#if __has_builtin(__builtin_amdgcn_tensor_load_to_lds) && \
    __has_builtin(__builtin_amdgcn_s_wait_tensorcnt)
#define HAVE_TDM 1
#else
#define HAVE_TDM 0
#endif

// ---------------------------------------------------------------------------
// Prep 1: fold embedding + input projection + bias into a per-token table:
//   P[v][g] = sum_e emb[v][e] * Wx[e][g] + b[g]      (VOCAB x 4U, per direction)
// ---------------------------------------------------------------------------
__global__ __launch_bounds__(256) void prep_P(
    const float* __restrict__ emb,
    const float* __restrict__ Wx_f, const float* __restrict__ b_f,
    const float* __restrict__ Wx_b, const float* __restrict__ b_b,
    float* __restrict__ Pf, float* __restrict__ Pb)
{
    int idx = blockIdx.x * blockDim.x + threadIdx.x;   // 2 * 128 * 1024
    int dir = idx >> 17;
    int rem = idx & 0x1FFFF;
    int v = rem >> 10;
    int g = rem & 1023;
    const float* Wx = dir ? Wx_b : Wx_f;
    const float* bs = dir ? b_b  : b_f;
    float acc = bs[g];
#pragma unroll 8
    for (int e = 0; e < EMB; ++e)
        acc = fmaf(emb[v * EMB + e], Wx[e * GATES + g], acc);
    (dir ? Pb : Pf)[v * GATES + g] = acc;
}

// ---------------------------------------------------------------------------
// Prep 2: pack Wh (fp32 [256,1024]) into f16 WMMA B fragments.
// Fragment (nt, ks): K = ks*32..+31, N = nt*16..+15; lane%16<->N, (lane/16, 16 halfs)<->K (table order).
// ---------------------------------------------------------------------------
__global__ __launch_bounds__(256) void pack_Wh(
    const float* __restrict__ Wh_f, const float* __restrict__ Wh_b,
    _Float16* __restrict__ Wp_f, _Float16* __restrict__ Wp_b)
{
    int idx = blockIdx.x * blockDim.x + threadIdx.x;   // 2 * 256 * 1024
    int dir  = idx >> 18;
    int rem  = idx & 0x3FFFF;
    int frag = rem >> 9;          // nt*8 + ks   (0..511)
    int lane = (rem >> 4) & 31;   // N within tile (lane%16), K half (lane/16)
    int j    = rem & 15;          // K element within lane half
    int nt = frag >> 3, ks = frag & 7;
    int k = ks * 32 + ((lane >> 4) << 3) + (j & 7) + ((j >> 3) << 4);
    int n = nt * 16 + (lane & 15);
    const float* Wh = dir ? Wh_b : Wh_f;
    (dir ? Wp_b : Wp_f)[rem] = (_Float16)Wh[k * GATES + n];
}

// ---------------------------------------------------------------------------
// Main scan. grid = (B/16, 2 directions), 256 threads = 8 waves.
// Workgroup owns 16 batch rows and ALL 1024 gate columns -> no grid syncs.
// Wave w owns units [32w, 32w+32): 8 N-tiles = {i,f,g,o} x {2 tiles}.
// ks-outer MMA loop: one live A fragment, 8 independent accumulators per
// issue group. WMMA chains start from inline C=0; P[token] added afterwards.
// Double-buffered h/token staging: one barrier per step.
// Wh staging into LDS uses the Tensor Data Mover when available.
// ---------------------------------------------------------------------------
__global__ __launch_bounds__(256) void lstm_scan(
    const int*      __restrict__ tokens,   // [B, SEQ]
    const float*    __restrict__ Pf, const float*    __restrict__ Pb,
    const _Float16* __restrict__ Wpf, const _Float16* __restrict__ Wpb,
    float* __restrict__ out, int B)
{
    const int dir    = blockIdx.y;
    const int bb     = blockIdx.x * MT;
    const int lane   = threadIdx.x & 31;
    const int w      = threadIdx.x >> 5;
    const int m16    = lane & 15;
    const int hiHalf = (lane >> 4) & 1;

    const float*    P  = dir ? Pb  : Pf;
    const _Float16* Wp = dir ? Wpb : Wpf;

    __shared__ _Float16 ldsW[NSTAGE_FRAGS * 512];  // 256 KB: Wh fragments, gates i,f
    __shared__ _Float16 hpack[2][MT * HP];         // 2 x 8.4 KB: h, A-fragment order
    __shared__ int tok[2][MT];

    // ---- stage gates i,f of packed Wh into LDS (one-time, 256 KB)
#if HAVE_TDM
    if (w == 0) {
        // TDM descriptor: 2D tensor/tile of 32768 x 1 elements, 8 B each.
        const uint32_t ldsoff = (uint32_t)(uintptr_t)(void*)&ldsW[0]; // LDS offset in addr[31:0]
        const uint64_t ga     = (uint64_t)(uintptr_t)Wp;
        const uint32_t D0     = (NSTAGE_FRAGS * 512 * 2) / 8;         // 32768 elements
        v4u g0;
        g0[0] = 1u;                                        // count=1, is_restore=0
        g0[1] = ldsoff;                                    // lds_addr
        g0[2] = (uint32_t)ga;                              // global_addr[31:0]
        g0[3] = ((uint32_t)(ga >> 32) & 0x01FFFFFFu) | (2u << 30);  // addr[56:32] | type=2
        v8i g1;
        g1[0] = 3 << 16;                                   // wg_mask=0, data_size=3 (8B)
        g1[1] = (int)((D0 & 0xFFFFu) << 16);               // tensor_dim0[15:0] @ bits 63:48
        g1[2] = (int)((D0 >> 16) | (1u << 16));            // dim0[31:16]; tensor_dim1=1
        g1[3] = (int)((D0 & 0xFFFFu) << 16);               // tile_dim0 @ bits 127:112
        g1[4] = 1;                                         // tile_dim1 = 1
        g1[5] = (int)D0;                                   // tensor_dim0_stride low 32
        g1[6] = 0;
        g1[7] = 0;
        v4i gz = {0, 0, 0, 0};
#if defined(__clang_major__) && (__clang_major__ >= 23)
        v8i gz8 = {0, 0, 0, 0, 0, 0, 0, 0};
        __builtin_amdgcn_tensor_load_to_lds(g0, g1, gz, gz, gz8, 0);
#else
        __builtin_amdgcn_tensor_load_to_lds(g0, g1, gz, gz, 0);
#endif
        __builtin_amdgcn_s_wait_tensorcnt(0);
    }
#else
    {
        const uint4* src = (const uint4*)Wp;
        uint4* dst = (uint4*)ldsW;
        for (int i = threadIdx.x; i < (NSTAGE_FRAGS * 512) / 8; i += blockDim.x)
            dst[i] = src[i];
    }
#endif
    for (int i = threadIdx.x; i < MT * HP; i += blockDim.x) hpack[0][i] = (_Float16)0.0f;
    if (threadIdx.x < MT) {
        int t0 = dir ? (SEQ - 1) : 0;
        tok[0][threadIdx.x] = tokens[(size_t)(bb + threadIdx.x) * SEQ + t0];
    }
    __syncthreads();

    float cst[2][8], hreg[2][8];
#pragma unroll
    for (int tI = 0; tI < 2; ++tI)
#pragma unroll
        for (int r = 0; r < 8; ++r) { cst[tI][r] = 0.0f; hreg[tI][r] = 0.0f; }

    union AF { v16h v; uint4 q[2]; };

    uintptr_t wbase = (uintptr_t)Wp;
    const v8f zero = {};

    for (int t = 0; t < SEQ; ++t) {
        const int cur = t & 1, nxt = cur ^ 1;
        // SGPR-launder the streamed-weight base: loop-variant from the
        // compiler's POV (no 256-VGPR hoist), stays scalar (saddr form).
        asm volatile("" : "+s"(wbase));
        const gf16* Wst = (const gf16*)wbase;

        // ---- z = h @ Wh (C starts at inline 0); ks outer so only one A
        // fragment is live and the 8 WMMAs per group are independent.
        v8f acc[8];
#pragma unroll
        for (int ks = 0; ks < 8; ++ks) {
            AF a;
            const int abase = m16 * HP + ks * 32 + hiHalf * 16;
            a.q[0] = *(const uint4*)&hpack[cur][abase];
            a.q[1] = *(const uint4*)&hpack[cur][abase + 8];
#pragma unroll
            for (int f = 0; f < 8; ++f) {
                const int gate = f >> 1, tI = f & 1;
                const int nt   = gate * 16 + w * 2 + tI;       // global N tile
                const size_t foff = ((size_t)(nt * 8 + ks) * 32 + lane) * 16;
                v16h bfr;
                if (gate < 2)   // compile-time after unroll: i,f from LDS
                    bfr = *(const v16h*)&ldsW[foff];
                else            // g,o streamed from L2 (laundered scalar base)
                    bfr = *(const gv16h*)(Wst + foff);
                acc[f] = __builtin_amdgcn_wmma_f32_16x16x32_f16(
                    false, a.v, false, bfr, (short)0,
                    (ks == 0) ? zero : acc[f], false, false);
            }
        }

        // ---- add xz_t = P[token] (bias folded in); short-lived load results
        const float* prow[8];
#pragma unroll
        for (int r = 0; r < 8; ++r)
            prow[r] = P + (size_t)tok[cur][r + (hiHalf ? 8 : 0)] * GATES + m16;
#pragma unroll
        for (int f = 0; f < 8; ++f) {
            const int colb = (f >> 1) * UNITS + w * 32 + (f & 1) * 16;
#pragma unroll
            for (int r = 0; r < 8; ++r)
                acc[f][r] += prow[r][colb];
        }

        // ---- gate math (i,f,g,o for a unit live in the same lane/reg slot)
#pragma unroll
        for (int tI = 0; tI < 2; ++tI) {
#pragma unroll
            for (int r = 0; r < 8; ++r) {
                float iv = acc[0 + tI][r];
                float fv = acc[2 + tI][r];
                float gv = acc[4 + tI][r];
                float ov = acc[6 + tI][r];
                float c = sig_fast(fv) * cst[tI][r] + sig_fast(iv) * tanh_fast(gv);
                float h = sig_fast(ov) * tanh_fast(c);
                cst[tI][r]  = c;
                hreg[tI][r] = h;
            }
        }

        // ---- publish h (f16) into the OTHER buffer, A-fragment-permuted:
        // unit u -> K bits: [4]->p-high, [3]->half-sel, [2:1]->p-low, [0]->pair
#pragma unroll
        for (int tI = 0; tI < 2; ++tI) {
            const int u  = w * 32 + tI * 16 + m16;
            const int r5 = u & 31;
            const int ks = u >> 5;
            const int p  = ((r5 >> 4) & 1) * 4 + ((r5 >> 1) & 3);
            const int hb = (r5 >> 3) & 1;
            const int hh = r5 & 1;
            const int pos = ks * 32 + hb * 16 + p * 2 + hh;
#pragma unroll
            for (int r = 0; r < 8; ++r) {
                int m = r + (hiHalf ? 8 : 0);
                hpack[nxt][m * HP + pos] = (_Float16)hreg[tI][r];
            }
        }
        if (threadIdx.x < MT && t + 1 < SEQ) {
            int tn = dir ? (SEQ - 2 - t) : (t + 1);
            tok[nxt][threadIdx.x] = tokens[(size_t)(bb + threadIdx.x) * SEQ + tn];
        }
        __syncthreads();   // nxt buffers complete; cur buffers free for reuse
    }

    // ---- final states: out=[B,512] concat, then fh, fc, bh, bc each [B,256]
    const size_t OFF_FH = (size_t)B * 512;
    const size_t OFF_FC = OFF_FH + (size_t)B * UNITS;
    const size_t OFF_BH = OFF_FC + (size_t)B * UNITS;
    const size_t OFF_BC = OFF_BH + (size_t)B * UNITS;
#pragma unroll
    for (int tI = 0; tI < 2; ++tI) {
#pragma unroll
        for (int r = 0; r < 8; ++r) {
            int m = r + (hiHalf ? 8 : 0);
            int u = w * 32 + tI * 16 + m16;
            size_t b = (size_t)(bb + m);
            float h = hreg[tI][r], c = cst[tI][r];
            if (dir == 0) {
                out[b * 512 + u]              = h;
                out[OFF_FH + b * UNITS + u]   = h;
                out[OFF_FC + b * UNITS + u]   = c;
            } else {
                out[b * 512 + 256 + u]        = h;
                out[OFF_BH + b * UNITS + u]   = h;
                out[OFF_BC + b * UNITS + u]   = c;
            }
        }
    }
}

// ---------------------------------------------------------------------------
extern "C" void kernel_launch(void* const* d_in, const int* in_sizes, int n_in,
                              void* d_out, int out_size, void* d_ws, size_t ws_size,
                              hipStream_t stream)
{
    const int*   inputs = (const int*)  d_in[0];
    const float* emb    = (const float*)d_in[1];
    const float* Wx_f   = (const float*)d_in[2];
    const float* Wh_f   = (const float*)d_in[3];
    const float* b_f    = (const float*)d_in[4];
    const float* Wx_b   = (const float*)d_in[5];
    const float* Wh_b   = (const float*)d_in[6];
    const float* b_b    = (const float*)d_in[7];
    const int B = in_sizes[0] / SEQ;

    char* ws = (char*)d_ws;
    float*    Pf  = (float*)(ws);                       // 512 KB
    float*    Pb  = (float*)(ws + (512u << 10));        // 512 KB
    _Float16* Wpf = (_Float16*)(ws + (1024u << 10));    // 512 KB
    _Float16* Wpb = (_Float16*)(ws + (1536u << 10));    // 512 KB

    prep_P <<<(2 * VOCAB * GATES) / 256, 256, 0, stream>>>(emb, Wx_f, b_f, Wx_b, b_b, Pf, Pb);
    pack_Wh<<<(2 * UNITS * GATES) / 256, 256, 0, stream>>>(Wh_f, Wh_b, Wpf, Wpb);

    dim3 grid(B / MT, 2);
    lstm_scan<<<grid, 256, 0, stream>>>(inputs, Pf, Pb, Wpf, Wpb, (float*)d_out, B);
}
